// Model_17918603559215
// MI455X (gfx1250) — matrix operand order repaired and checked by reading, not compiled, not run on hardware
//
#include <hip/hip_runtime.h>
#include <hip/hip_bf16.h>
#include <math.h>

typedef __attribute__((ext_vector_type(16))) _Float16 v16h;
typedef __attribute__((ext_vector_type(8)))  float    v8f;

#define NEG_INF (-__builtin_inff())

// ---------------------------------------------------------------------------
// 1) Series decomposition (moving avg k=25, edge-replicated) + DLinear heads.
// ---------------------------------------------------------------------------
__global__ void k_decomp_dlinear(const float* __restrict__ xe,
                                 const float* __restrict__ Ws, const float* __restrict__ bs,
                                 const float* __restrict__ Wt, const float* __restrict__ bt,
                                 float* __restrict__ xP)
{
  __shared__ float xr[512];
  __shared__ float mm[512];
  const int bc  = blockIdx.x;          // b*862 + c
  const int c   = bc % 862;
  const int bb  = bc / 862;
  const int tid = threadIdx.x;         // 0..95
  for (int l = tid; l < 512; l += 96)
    xr[l] = xe[((size_t)bb * 512 + l) * 862 + c];
  __syncthreads();
  for (int l = tid; l < 512; l += 96) {
    float a = 0.f;
    #pragma unroll
    for (int j = -12; j <= 12; ++j) {
      int p = l + j; p = p < 0 ? 0 : (p > 511 ? 511 : p);
      a += xr[p];
    }
    mm[l] = a * (1.0f / 25.0f);
  }
  __syncthreads();
  float as = 0.f, at = 0.f;
  for (int l = 0; l < 512; ++l) {
    float m = mm[l];
    float r = xr[l] - m;
    as += r * Ws[(size_t)tid * 512 + l];
    at += m * Wt[(size_t)tid * 512 + l];
  }
  xP[(size_t)bc * 96 + tid] = as + bs[tid] + at + bt[tid];
}

// ---------------------------------------------------------------------------
// 2) WMMA GEMM: C[M, N] (+)= A[M,K] * B + bias, wave tile = 16 x (16*NT).
//    REQUIRES M%16==0, N%(16*NT)==0, K%32==0 (true for every call here).
//    One wave stages a 16x32 A tile once per k-step and reuses its A fragment
//    across NT back-to-back independent v_wmma_f32_16x16x32_f16 issues.
//    bkn==1 : B stored [K,N] row-major (ldb=N)   (Wup, Wq/Wk/Wv, Wdn)
//    bkn==0 : B stored [N,K] row-major (ldb=K)   (Wp1, Wp2 : weight-transposed)
//    beta==1: accumulate into existing C (residual add).
// ---------------------------------------------------------------------------
template <int NT>
__global__ __launch_bounds__(32, 1)
void k_gemm_wmma(const float* __restrict__ A, int lda,
                 const float* __restrict__ Bm, int ldb, int bkn,
                 const float* __restrict__ bias,
                 float* __restrict__ C, int ldc,
                 int M, int N, int K, int beta)
{
  __shared__ _Float16 As[16 * 32];        // [row][k]        (k contiguous)
  __shared__ _Float16 Bs[NT * 16 * 32];   // NT x [n][k]     (k contiguous)
  const int lane = threadIdx.x & 31;
  const int half = lane >> 4, l15 = lane & 15;
  const int m0 = blockIdx.y * 16, n0 = blockIdx.x * (16 * NT);
  const int hrow = lane >> 1;             // 0..15 (half-row owner)
  const int hcol = (lane & 1) * 16;       // 0 or 16
  v8f acc[NT];
  const v8f vzero = {};
  #pragma unroll
  for (int t = 0; t < NT; ++t) acc[t] = vzero;

  for (int kb = 0; kb < K; kb += 32) {
    // ---- stage A tile: lane loads 16 contiguous floats (4 x b128) ----
    {
      const float4* ap = (const float4*)(A + (size_t)(m0 + hrow) * lda + kb + hcol);
      #pragma unroll
      for (int u = 0; u < 4; ++u) {
        const float4 f = ap[u];
        const int o = hrow * 32 + hcol + u * 4;
        As[o + 0] = (_Float16)f.x; As[o + 1] = (_Float16)f.y;
        As[o + 2] = (_Float16)f.z; As[o + 3] = (_Float16)f.w;
      }
    }
    // ---- stage NT B sub-tiles into [n][k] order ----
    if (bkn) {
      const int krow = lane;              // 0..31
      const float4* bp = (const float4*)(Bm + (size_t)(kb + krow) * ldb + n0);
      #pragma unroll
      for (int t = 0; t < NT; ++t) {
        #pragma unroll
        for (int u = 0; u < 4; ++u) {
          const float4 f = bp[t * 4 + u];
          const int nb = u * 4;
          _Float16* bt_ = Bs + t * 512;
          bt_[(nb + 0) * 32 + krow] = (_Float16)f.x;
          bt_[(nb + 1) * 32 + krow] = (_Float16)f.y;
          bt_[(nb + 2) * 32 + krow] = (_Float16)f.z;
          bt_[(nb + 3) * 32 + krow] = (_Float16)f.w;
        }
      }
    } else {
      #pragma unroll
      for (int t = 0; t < NT; ++t) {
        const float4* bp =
            (const float4*)(Bm + (size_t)(n0 + t * 16 + hrow) * ldb + kb + hcol);
        #pragma unroll
        for (int u = 0; u < 4; ++u) {
          const float4 f = bp[u];
          const int o = t * 512 + hrow * 32 + hcol + u * 4;
          Bs[o + 0] = (_Float16)f.x; Bs[o + 1] = (_Float16)f.y;
          Bs[o + 2] = (_Float16)f.z; Bs[o + 3] = (_Float16)f.w;
        }
      }
    }
    __syncthreads();
    // ---- A fragment once, NT B fragments + WMMAs ----
    v16h afr;
    #pragma unroll
    for (int e = 0; e < 16; ++e) {
      const int v = e >> 1, p = e & 1;
      const int kk = ((v < 4) ? (v * 2) : (16 + (v - 4) * 2)) + half * 8 + p;
      afr[e] = As[l15 * 32 + kk];                 // two 16B runs per lane
    }
    #pragma unroll
    for (int t = 0; t < NT; ++t) {
      v16h bfr;
      #pragma unroll
      for (int e = 0; e < 16; ++e)
        bfr[e] = Bs[t * 512 + l15 * 32 + ((e < 8) ? (half * 8 + e) : (16 + half * 8 + e - 8))];   // two 16B runs
      acc[t] = __builtin_amdgcn_wmma_f32_16x16x32_f16(false, afr, false, bfr,
                                                      (short)0, acc[t], false, false);
    }
    __syncthreads();
  }
  #pragma unroll
  for (int t = 0; t < NT; ++t) {
    #pragma unroll
    for (int i = 0; i < 8; ++i) {
      const int r  = m0 + i + half * 8;
      const int cn = n0 + t * 16 + l15;
      float v = acc[t][i];
      if (bias) v += bias[cn];
      const size_t idx = (size_t)r * ldc + cn;
      if (beta) v += C[idx];
      C[idx] = v;
    }
  }
}

// ---------------------------------------------------------------------------
// 3) Pre-LN / post-LN (weight-only). One block (D threads) per row.
// ---------------------------------------------------------------------------
__global__ void k_layernorm(const float* __restrict__ x, const float* __restrict__ g,
                            float* __restrict__ y, int D)
{
  const int row = blockIdx.x;
  const int tid = threadIdx.x;
  float v  = x[(size_t)row * D + tid];
  float s = v, s2 = v * v;
  #pragma unroll
  for (int off = 16; off; off >>= 1) {
    s  += __shfl_xor(s,  off, 32);
    s2 += __shfl_xor(s2, off, 32);
  }
  __shared__ float sh1[8], sh2[8];
  const int w = tid >> 5;
  if ((tid & 31) == 0) { sh1[w] = s; sh2[w] = s2; }
  __syncthreads();
  if (tid == 0) {
    const int nw = blockDim.x >> 5;
    float a = 0.f, b = 0.f;
    for (int i = 0; i < nw; ++i) { a += sh1[i]; b += sh2[i]; }
    sh1[0] = a; sh2[0] = b;
  }
  __syncthreads();
  const float mean = sh1[0] / (float)D;
  const float var  = sh2[0] / (float)D - mean * mean;
  y[(size_t)row * D + tid] = (v - mean) * rsqrtf(var + 1e-5f) * g[tid];
}

// ---------------------------------------------------------------------------
// 4) Causal depthwise conv (k=4, zero left pad) + SiLU on x_in = up[:, :256].
// ---------------------------------------------------------------------------
__global__ void k_conv_silu(const float* __restrict__ up, const float* __restrict__ wc,
                            const float* __restrict__ bc, float* __restrict__ xc,
                            int S)
{
  const int idx = blockIdx.x * blockDim.x + threadIdx.x;
  if (idx >= 8 * S * 256) return;
  const int ch = idx & 255;
  const int rs = idx >> 8;              // b*S + s
  const int s  = rs % S;
  const int bb = rs / S;
  float acc = bc[ch];
  #pragma unroll
  for (int j = 0; j < 4; ++j) {
    const int sp = s - 3 + j;
    if (sp >= 0) acc += up[((size_t)(bb * S + sp)) * 512 + ch] * wc[j * 256 + ch];
  }
  xc[(size_t)rs * 256 + ch] = acc / (1.f + __expf(-acc));
}

// ---------------------------------------------------------------------------
// 5) Gate preactivations: ig = g_in@Wi+bi, logf = log_sigmoid(g_in@Wf+bf).
// ---------------------------------------------------------------------------
__global__ void k_gates(const float* __restrict__ q, const float* __restrict__ k,
                        const float* __restrict__ v,
                        const float* __restrict__ Wi, const float* __restrict__ bi,
                        const float* __restrict__ Wf, const float* __restrict__ bfv,
                        float* __restrict__ igv, float* __restrict__ logf, int S)
{
  const int idx = blockIdx.x * blockDim.x + threadIdx.x;
  if (idx >= 8 * 4 * S) return;
  const int s  = idx % S;
  const int bh = idx / S;
  const int h  = bh & 3;
  const int bb = bh >> 2;
  const size_t row = (size_t)bb * S + s;
  float ai = bi[h], af = bfv[h];
  for (int j = 0; j < 256; ++j) {
    const float g0 = q[row * 256 + j];
    const float g1 = k[row * 256 + j];
    const float g2 = v[row * 256 + j];
    ai += g0 * Wi[j * 4 + h] + g1 * Wi[(256 + j) * 4 + h] + g2 * Wi[(512 + j) * 4 + h];
    af += g0 * Wf[j * 4 + h] + g1 * Wf[(256 + j) * 4 + h] + g2 * Wf[(512 + j) * 4 + h];
  }
  igv[idx]  = ai;
  logf[idx] = fminf(af, 0.f) - log1pf(__expf(-fabsf(af)));   // log sigmoid
}

__global__ void k_cumsum(const float* __restrict__ logf, float* __restrict__ cum, int S)
{
  if (threadIdx.x != 0) return;
  const size_t base = (size_t)blockIdx.x * S;
  float r = 0.f;
  for (int s = 0; s < S; ++s) { r += logf[base + s]; cum[base + s] = r; }
}

// ---------------------------------------------------------------------------
// 6) Flash-style decay attention. grid = (ceil(S/16), NH=4, B=8), block = 32
//    (one wave, registers sized for 1 resident wave -> no spills).
//    m is only an overflow guard (result is m-invariant), so row-max uses
//    shuffles but the row-sum for the |sum| normalizer is computed on the
//    matrix unit: P(16x32) @ ones(32x16) -> every lane holds its row's sum.
// ---------------------------------------------------------------------------
__global__ __launch_bounds__(32, 1)
void k_attn(const float* __restrict__ qg, const float* __restrict__ kg,
            const float* __restrict__ vg,
            const float* __restrict__ cum, const float* __restrict__ igp,
            float* __restrict__ hb, int S)
{
  __shared__ _Float16 qs[16 * 64];
  __shared__ _Float16 ks[32 * 64];
  __shared__ _Float16 vs[32 * 64];
  __shared__ _Float16 ps[16 * 32];
  const int lane = threadIdx.x;
  const int half = lane >> 4, l15 = lane & 15;
  const int t0 = blockIdx.x * 16;
  const int h  = blockIdx.y, b = blockIdx.z;
  const size_t rowbase = (size_t)b * S;
  const int cbase = h * 64;
  const float kscale = 0.125f;          // DH^-0.5, DH=64

  // stage q tile: 64 segments of 16 floats, float4 loads
  for (int seg = lane; seg < 64; seg += 32) {
    const int r = seg >> 2, q4 = (seg & 3) * 16;
    const int t = t0 + r;
    const int o = r * 64 + q4;
    if (t < S) {
      const float4* p = (const float4*)(qg + (rowbase + t) * 256 + cbase + q4);
      #pragma unroll
      for (int u = 0; u < 4; ++u) {
        const float4 f = p[u];
        qs[o + u * 4 + 0] = (_Float16)f.x; qs[o + u * 4 + 1] = (_Float16)f.y;
        qs[o + u * 4 + 2] = (_Float16)f.z; qs[o + u * 4 + 3] = (_Float16)f.w;
      }
    } else {
      #pragma unroll
      for (int u = 0; u < 16; ++u) qs[o + u] = (_Float16)0.f;
    }
  }
  __syncthreads();
  // q A-fragments for K (=dh) 0..31 and 32..63, loaded once
  v16h a0, a1;
  #pragma unroll
  for (int e = 0; e < 16; ++e) {
    const int v = e >> 1, p = e & 1;
    const int kk = ((v < 4) ? (v * 2) : (16 + (v - 4) * 2)) + half * 8 + p;
    a0[e] = qs[l15 * 64 + kk];
    a1[e] = qs[l15 * 64 + 32 + kk];
  }
  // constant all-ones B fragment for row-sum WMMA
  v16h ones;
  #pragma unroll
  for (int e = 0; e < 16; ++e) ones[e] = (_Float16)1.0f;

  const size_t gbase = ((size_t)b * 4 + h) * S;
  float cum_t[8], mrow[8], lrow[8];
  v8f hacc[4];
  const v8f vzero = {};
  #pragma unroll
  for (int i = 0; i < 8; ++i) {
    const int t = t0 + half * 8 + i;
    cum_t[i] = (t < S) ? cum[gbase + t] : 0.f;
    mrow[i] = NEG_INF; lrow[i] = 0.f;
  }
  #pragma unroll
  for (int j = 0; j < 4; ++j) hacc[j] = vzero;

  int t_hi = t0 + 15; if (t_hi >= S) t_hi = S - 1;
  for (int k0 = 0; k0 <= t_hi; k0 += 32) {
    // stage k (scaled) and v tiles: 128 segments of 16 floats each
    for (int seg = lane; seg < 128; seg += 32) {
      const int r = seg >> 2, q4 = (seg & 3) * 16;
      const int s = k0 + r;
      const int o = r * 64 + q4;
      if (s < S) {
        const size_t gofs = (rowbase + s) * 256 + cbase + q4;
        const float4* kp = (const float4*)(kg + gofs);
        const float4* vp = (const float4*)(vg + gofs);
        #pragma unroll
        for (int u = 0; u < 4; ++u) {
          const float4 fk = kp[u];
          const float4 fv = vp[u];
          ks[o + u * 4 + 0] = (_Float16)(fk.x * kscale);
          ks[o + u * 4 + 1] = (_Float16)(fk.y * kscale);
          ks[o + u * 4 + 2] = (_Float16)(fk.z * kscale);
          ks[o + u * 4 + 3] = (_Float16)(fk.w * kscale);
          vs[o + u * 4 + 0] = (_Float16)fv.x; vs[o + u * 4 + 1] = (_Float16)fv.y;
          vs[o + u * 4 + 2] = (_Float16)fv.z; vs[o + u * 4 + 3] = (_Float16)fv.w;
        }
      } else {
        #pragma unroll
        for (int u = 0; u < 16; ++u) { ks[o + u] = (_Float16)0.f; vs[o + u] = (_Float16)0.f; }
      }
    }
    __syncthreads();
    // raw scores q.k^T for two 16-key subtiles
    v8f sa0, sa1;
    float cum_s0, cum_s1, ig_s0, ig_s1;
    {
      const int s0 = k0 + l15;
      const int s1 = k0 + 16 + l15;
      cum_s0 = (s0 < S) ? cum[gbase + s0] : 0.f;
      ig_s0  = (s0 < S) ? igp[gbase + s0] : 0.f;
      cum_s1 = (s1 < S) ? cum[gbase + s1] : 0.f;
      ig_s1  = (s1 < S) ? igp[gbase + s1] : 0.f;
    }
    #pragma unroll
    for (int st = 0; st < 2; ++st) {
      v16h b0, b1;
      #pragma unroll
      for (int e = 0; e < 16; ++e) {
        b0[e] = ks[(st * 16 + l15) * 64 + ((e < 8) ? (half * 8 + e) : (16 + half * 8 + e - 8))];
        b1[e] = ks[(st * 16 + l15) * 64 + 32 + ((e < 8) ? (half * 8 + e) : (16 + half * 8 + e - 8))];
      }
      v8f z = vzero;
      z = __builtin_amdgcn_wmma_f32_16x16x32_f16(false, a0, false, b0, (short)0, z, false, false);
      z = __builtin_amdgcn_wmma_f32_16x16x32_f16(false, a1, false, b1, (short)0, z, false, false);
      if (st == 0) sa0 = z; else sa1 = z;
    }
    // fused per-row pass: decay logD, overflow-guard row max (exact result is
    // invariant to m), rescale, weights -> LDS
    #pragma unroll
    for (int i = 0; i < 8; ++i) {
      const int t = t0 + half * 8 + i;
      const int s0 = k0 + l15;
      const int s1 = k0 + 16 + l15;
      float d0 = (t < S && s0 <= t) ? (cum_t[i] - cum_s0 + ig_s0) : NEG_INF;
      float d1 = (t < S && s1 <= t) ? (cum_t[i] - cum_s1 + ig_s1) : NEG_INF;
      float dm = fmaxf(d0, d1);
      for (int off = 1; off < 16; off <<= 1)
        dm = fmaxf(dm, __shfl_xor(dm, off, 32));   // stays within 16-lane half
      const float mnew = fmaxf(mrow[i], dm);
      const float sc = (mnew == NEG_INF) ? 1.f : __expf(mrow[i] - mnew);
      mrow[i] = mnew;
      lrow[i] *= sc;
      #pragma unroll
      for (int j = 0; j < 4; ++j) hacc[j][i] *= sc;
      const float p0 = (d0 == NEG_INF) ? 0.f : sa0[i] * __expf(d0 - mnew);
      const float p1 = (d1 == NEG_INF) ? 0.f : sa1[i] * __expf(d1 - mnew);
      const int r = i + half * 8;
      ps[r * 32 + l15]      = (_Float16)p0;
      ps[r * 32 + 16 + l15] = (_Float16)p1;
    }
    __syncthreads();
    v16h ap;
    #pragma unroll
    for (int e = 0; e < 16; ++e) {
      const int v = e >> 1, p = e & 1;
      const int kk = ((v < 4) ? (v * 2) : (16 + (v - 4) * 2)) + half * 8 + p;
      ap[e] = ps[l15 * 32 + kk];
    }
    // row sums on the matrix unit: every column of P @ ones = row sum
    {
      v8f lsum = vzero;
      lsum = __builtin_amdgcn_wmma_f32_16x16x32_f16(false, ap, false, ones,
                                                    (short)0, lsum, false, false);
      #pragma unroll
      for (int i = 0; i < 8; ++i) lrow[i] += lsum[i];
    }
    #pragma unroll
    for (int j = 0; j < 4; ++j) {
      v16h bv;
      #pragma unroll
      for (int e = 0; e < 16; ++e)
        bv[e] = vs[((e < 8) ? (half * 8 + e) : (16 + half * 8 + e - 8)) * 64 + j * 16 + l15];
      hacc[j] = __builtin_amdgcn_wmma_f32_16x16x32_f16(false, ap, false, bv,
                                                       (short)0, hacc[j], false, false);
    }
    __syncthreads();
  }
  // normalize: norm = max(|sum scores|, exp(-m))
  #pragma unroll
  for (int i = 0; i < 8; ++i) {
    const int t = t0 + half * 8 + i;
    if (t >= S) continue;
    const float norm = fmaxf(fabsf(lrow[i]), __expf(-mrow[i]));
    const float inv = 1.f / norm;
    #pragma unroll
    for (int j = 0; j < 4; ++j)
      hb[(rowbase + t) * 256 + cbase + j * 16 + l15] = hacc[j][i] * inv;
  }
}

// ---------------------------------------------------------------------------
// 7) Per-head LN (weight only) + (hn + skip*xc) * silu(z). Block=64 per (row,head).
// ---------------------------------------------------------------------------
__global__ void k_mhln_mix(const float* __restrict__ h, const float* __restrict__ up,
                           const float* __restrict__ xc, const float* __restrict__ mhg,
                           const float* __restrict__ skip, float* __restrict__ hs)
{
  const int row = blockIdx.x, head = blockIdx.y, tid = threadIdx.x; // 64 threads
  const int c = head * 64 + tid;
  const float v = h[(size_t)row * 256 + c];
  float s = v, s2 = v * v;
  #pragma unroll
  for (int off = 16; off; off >>= 1) {
    s  += __shfl_xor(s,  off, 32);
    s2 += __shfl_xor(s2, off, 32);
  }
  __shared__ float p1[2], p2[2];
  if ((tid & 31) == 0) { p1[tid >> 5] = s; p2[tid >> 5] = s2; }
  __syncthreads();
  const float S1 = p1[0] + p1[1], S2 = p2[0] + p2[1];
  const float mean = S1 * (1.f / 64.f);
  const float var  = S2 * (1.f / 64.f) - mean * mean;
  const float hn = (v - mean) * rsqrtf(var + 1e-5f) * mhg[c];
  const float z  = up[(size_t)row * 512 + 256 + c];
  const float sz = z / (1.f + __expf(-z));
  hs[(size_t)row * 256 + c] = (hn + skip[c] * xc[(size_t)row * 256 + c]) * sz;
}

// ---------------------------------------------------------------------------
// 8) [B,C,P] -> [B,P,C]
// ---------------------------------------------------------------------------
__global__ void k_transpose_out(const float* __restrict__ t, float* __restrict__ out)
{
  const int idx = blockIdx.x * blockDim.x + threadIdx.x;
  if (idx >= 8 * 96 * 862) return;
  const int c   = idx % 862;
  const int rem = idx / 862;
  const int p   = rem % 96;
  const int bb  = rem / 96;
  out[idx] = t[((size_t)bb * 862 + c) * 96 + p];
}

// ---------------------------------------------------------------------------
extern "C" void kernel_launch(void* const* d_in, const int* in_sizes, int n_in,
                              void* d_out, int out_size, void* d_ws, size_t ws_size,
                              hipStream_t stream)
{
  (void)in_sizes; (void)n_in; (void)out_size; (void)ws_size;
  const float* x_enc  = (const float*)d_in[0];
  const float* Ws     = (const float*)d_in[4];
  const float* bs     = (const float*)d_in[5];
  const float* Wt     = (const float*)d_in[6];
  const float* bt     = (const float*)d_in[7];
  const float* Wp1    = (const float*)d_in[8];
  const float* bp1    = (const float*)d_in[9];
  const float* Wp2    = (const float*)d_in[10];
  const float* bp2    = (const float*)d_in[11];
  const float* ln_g   = (const float*)d_in[12];
  const float* Wup    = (const float*)d_in[13];
  const float* bup    = (const float*)d_in[14];
  const float* Wconv  = (const float*)d_in[15];
  const float* bconv  = (const float*)d_in[16];
  const float* Wq     = (const float*)d_in[17];
  const float* Wk     = (const float*)d_in[18];
  const float* Wv     = (const float*)d_in[19];
  const float* Wi     = (const float*)d_in[20];
  const float* bi     = (const float*)d_in[21];
  const float* Wf     = (const float*)d_in[22];
  const float* bfp    = (const float*)d_in[23];
  const float* mh_g   = (const float*)d_in[24];
  const float* skip   = (const float*)d_in[25];
  const float* Wdn    = (const float*)d_in[26];
  const float* bdn    = (const float*)d_in[27];
  const float* post_g = (const float*)d_in[28];

  const int S = 862;
  const int M = 8 * S;                        // 6896 tokens (431 * 16)
  float* ws = (float*)d_ws;
  size_t off = 0;
  auto alloc = [&](size_t n) { float* p = ws + off; off += n; return p; };
  float* xP   = alloc((size_t)M * 96);
  float* x0   = alloc((size_t)M * 128);
  float* xn   = alloc((size_t)M * 128);
  float* yln  = alloc((size_t)M * 128);
  float* up   = alloc((size_t)M * 512);
  float* xc   = alloc((size_t)M * 256);
  float* qb   = alloc((size_t)M * 256);
  float* kb   = alloc((size_t)M * 256);
  float* vb   = alloc((size_t)M * 256);
  float* hbuf = alloc((size_t)M * 256);
  float* hsb  = alloc((size_t)M * 256);
  float* igv  = alloc((size_t)8 * 4 * S);
  float* lgf  = alloc((size_t)8 * 4 * S);
  float* cum  = alloc((size_t)8 * 4 * S);

  k_decomp_dlinear<<<M, 96, 0, stream>>>(x_enc, Ws, bs, Wt, bt, xP);

  // Wp1: N=128 -> 2 wave-strips of 64
  dim3 g1(128 / 64, M / 16);
  k_gemm_wmma<4><<<g1, 32, 0, stream>>>(xP, 96, Wp1, 96, 0, bp1, x0, 128, M, 128, 96, 0);

  float* cur = x0; float* nxt = xn;
  for (int blk = 0; blk < 2; ++blk) {
    k_layernorm<<<M, 128, 0, stream>>>(cur, ln_g + blk * 128, yln, 128);

    dim3 gu(512 / 64, M / 16);
    k_gemm_wmma<4><<<gu, 32, 0, stream>>>(yln, 128, Wup + (size_t)blk * 128 * 512, 512, 1,
                                          bup + blk * 512, up, 512, M, 512, 128, 0);

    const int tc = M * 256;
    k_conv_silu<<<(tc + 255) / 256, 256, 0, stream>>>(up, Wconv + blk * 4 * 256,
                                                      bconv + blk * 256, xc, S);

    dim3 gq(256 / 64, M / 16);
    k_gemm_wmma<4><<<gq, 32, 0, stream>>>(xc, 256, Wq + (size_t)blk * 65536, 256, 1,
                                          nullptr, qb, 256, M, 256, 256, 0);
    k_gemm_wmma<4><<<gq, 32, 0, stream>>>(xc, 256, Wk + (size_t)blk * 65536, 256, 1,
                                          nullptr, kb, 256, M, 256, 256, 0);
    k_gemm_wmma<4><<<gq, 32, 0, stream>>>(up, 512, Wv + (size_t)blk * 65536, 256, 1,
                                          nullptr, vb, 256, M, 256, 256, 0);

    const int ng = 8 * 4 * S;
    k_gates<<<(ng + 127) / 128, 128, 0, stream>>>(qb, kb, vb,
                                                  Wi + blk * 768 * 4, bi + blk * 4,
                                                  Wf + blk * 768 * 4, bfp + blk * 4,
                                                  igv, lgf, S);
    k_cumsum<<<32, 32, 0, stream>>>(lgf, cum, S);

    dim3 ga((S + 15) / 16, 4, 8);
    k_attn<<<ga, 32, 0, stream>>>(qb, kb, vb, cum, igv, hbuf, S);

    dim3 gm(M, 4);
    k_mhln_mix<<<gm, 64, 0, stream>>>(hbuf, up, xc, mh_g + blk * 256,
                                      skip + blk * 256, hsb);

    hipMemcpyAsync(nxt, cur, (size_t)M * 128 * sizeof(float),
                   hipMemcpyDeviceToDevice, stream);
    dim3 gd(128 / 64, M / 16);
    k_gemm_wmma<4><<<gd, 32, 0, stream>>>(hsb, 256, Wdn + (size_t)blk * 256 * 128, 128, 1,
                                          bdn + blk * 128, nxt, 128, M, 128, 256, 1);
    float* t = cur; cur = nxt; nxt = t;
  }

  k_layernorm<<<M, 128, 0, stream>>>(cur, post_g, yln, 128);
  // Wp2: N=96 -> 3 wave-strips of 32 (NT=2)
  dim3 g2(96 / 32, M / 16);
  k_gemm_wmma<2><<<g2, 32, 0, stream>>>(yln, 128, Wp2, 128, 0, bp2, xP, 96, M, 96, 128, 0);

  const int no = 8 * 96 * 862;
  k_transpose_out<<<(no + 255) / 256, 256, 0, stream>>>(xP, (float*)d_out);
}
